// InfAffine_36352603193790
// MI455X (gfx1250) — matrix operand order repaired and checked by reading, not compiled, not run on hardware
//
#include <hip/hip_runtime.h>

// ---------------------------------------------------------------------------
// Fused: mus column-normalize + exp(-10*alpha) precompute
//        -> f16 WMMA GEMM (v_wmma_f32_16x16x32_f16), 2 interleaved acc chains,
//           B double-buffered -> raw-instruction softmin epilogue -> softplus
// ---------------------------------------------------------------------------

typedef _Float16 half16 __attribute__((ext_vector_type(16)));
typedef float    v8f    __attribute__((ext_vector_type(8)));

#define DDIM 128    // feature dim (GEMM reduction dim)
#define KPROTO 1024 // number of prototypes (GEMM N dim, softmin dim)

#if __has_builtin(__builtin_amdgcn_sqrtf)
#define RAW_SQRTF(x) __builtin_amdgcn_sqrtf(x)
#else
#define RAW_SQRTF(x) sqrtf(x)
#endif
#if __has_builtin(__builtin_amdgcn_exp2f)
#define RAW_EXP2F(x) __builtin_amdgcn_exp2f(x)
#else
#define RAW_EXP2F(x) exp2f(x)
#endif
#if __has_builtin(__builtin_amdgcn_fmed3f)
#define CLAMP3(x, lo, hi) __builtin_amdgcn_fmed3f((x), (lo), (hi))
#else
#define CLAMP3(x, lo, hi) fminf(fmaxf((x), (lo)), (hi))
#endif

// --- Kernel 1: normalize columns of mus [D,K] -> transposed f16 [K,D],
//               and precompute ea[k] = exp(-alphas[k]/gamma).
__global__ __launch_bounds__(256) void prep_kernel(
    const float* __restrict__ mus, const float* __restrict__ alphas,
    _Float16* __restrict__ bt, float* __restrict__ ea) {
  const int k = blockIdx.x * blockDim.x + threadIdx.x;
  if (k >= KPROTO) return;
  float ss = 0.0f;
#pragma unroll 8
  for (int d = 0; d < DDIM; ++d) {
    float v = mus[(size_t)d * KPROTO + k];
    ss += v * v;
  }
  const float r = rsqrtf(ss);
#pragma unroll 8
  for (int d = 0; d < DDIM; ++d) {
    bt[(size_t)k * DDIM + d] = (_Float16)(mus[(size_t)d * KPROTO + k] * r);
  }
  ea[k] = __expf(-10.0f * alphas[k]);
}

// --- Kernel 2: fused WMMA GEMM + transcendental epilogue + softmin reduce.
// One wave owns 16 rows of xs; sweeps 1024 prototype columns as 64 16-wide
// tiles, two tiles in flight (interleaved WMMA chains + pipelined B loads).
__global__ __launch_bounds__(256) void softmin_wmma(
    const float* __restrict__ xs,      // [B,128] f32, rows unit-norm
    const _Float16* __restrict__ bt,   // [1024,128] f16, normalized prototypes
    const float* __restrict__ ea,      // [1024] exp(-10*alpha)
    float* __restrict__ out) {         // [B]
  const int lane = threadIdx.x & 31;
  const int wave = threadIdx.x >> 5;
  const int ln   = lane & 15;
  const bool hi  = (lane & 16) != 0;
  const int m0   = (blockIdx.x * 8 + wave) * 16;

  // ---- A fragments: 16 rows x 128 K, f32 -> f16, resident in VGPRs.
  half16 a[4];
  const float* arow = xs + (size_t)(m0 + ln) * DDIM;
#pragma unroll
  for (int ks = 0; ks < 4; ++ks) {
    const int base = ks * 32 + (hi ? 16 : 0);
    const float4 f0 = *(const float4*)(arow + base);
    const float4 f1 = *(const float4*)(arow + base + 4);
    const float4 f2 = *(const float4*)(arow + base + 8);
    const float4 f3 = *(const float4*)(arow + base + 12);
    const float t[16] = {f0.x, f0.y, f0.z, f0.w, f1.x, f1.y, f1.z, f1.w,
                         f2.x, f2.y, f2.z, f2.w, f3.x, f3.y, f3.z, f3.w};
#pragma unroll
    for (int i = 0; i < 16; ++i) a[ks][i] = (_Float16)t[i];
  }

  float S[8];
#pragma unroll
  for (int r = 0; r < 8; ++r) S[r] = 0.0f;

  auto loadB = [&](int tile, half16 b[4]) {
    const _Float16* brow =
        bt + (size_t)(tile * 16 + ln) * DDIM + (hi ? 16 : 0);
#pragma unroll
    for (int ks = 0; ks < 4; ++ks) b[ks] = *(const half16*)(brow + ks * 32);
  };

  // exp(-cost/0.1) = exp2(acos(ip)^2 * -5*log2e) * exp(-10*alpha)
  // Arg ranges are bounded => raw v_sqrt/v_exp (no libm denorm fixups).
  auto epilog = [&](const v8f& acc, float e) {
#pragma unroll
    for (int r = 0; r < 8; ++r) {
      const float ip = CLAMP3(acc[r], -0.99999988f, 0.99999988f);
      const float ax = fabsf(ip);
      const float P = 1.5707288f +
                      ax * (-0.2121144f + ax * (0.0742610f - 0.0187293f * ax));
      float th = RAW_SQRTF(1.0f - ax) * P;       // acos(|ip|), abs err < 7e-5
      th = (ip < 0.0f) ? (3.14159265358979f - th) : th;
      const float w = RAW_EXP2F(th * th * -7.2134752f);  // exp(-5*th^2)
      S[r] = fmaf(e, w, S[r]);
    }
  };

  half16 b0[4], b1[4];
  loadB(0, b0);
  loadB(1, b1);

  for (int t = 0; t < 64; t += 2) {
    __builtin_prefetch(bt + (size_t)((((t + 6) & 63) * 16) + ln) * DDIM, 0, 1);
    const float e0 = ea[t * 16 + ln];
    const float e1 = ea[(t + 1) * 16 + ln];

    // Two interleaved accumulator chains: each WMMA's RAW dependency is two
    // instructions back, hiding XDL latency without v_nop padding.
    v8f acc0 = {}, acc1 = {};
#pragma unroll
    for (int ks = 0; ks < 4; ++ks) {
      acc0 = __builtin_amdgcn_wmma_f32_16x16x32_f16(
          false, a[ks], false, b0[ks], (short)0, acc0, false, false);
      acc1 = __builtin_amdgcn_wmma_f32_16x16x32_f16(
          false, a[ks], false, b1[ks], (short)0, acc1, false, false);
    }

    // B consumed: issue next pair now so L2 latency overlaps both epilogues.
    loadB((t + 2) & 63, b0);  // wraps harmlessly on the final iteration
    loadB((t + 3) & 63, b1);

    epilog(acc0, e0);
    epilog(acc1, e1);
  }

  // Sum partial exp-sums across the 16 lanes of each half-wave.
#pragma unroll
  for (int m = 1; m <= 8; m <<= 1) {
#pragma unroll
    for (int r = 0; r < 8; ++r) S[r] += __shfl_xor(S[r], m, 32);
  }

  if (ln == 0) {
    const int rbase = m0 + (hi ? 8 : 0);
#pragma unroll
    for (int r = 0; r < 8; ++r) {
      // F = 0.1*log(S); out = 0.1*log1p(exp(-F/0.1)) = 0.1*log1p(1/S)
      out[rbase + r] = 0.1f * log1pf(1.0f / S[r]);
    }
  }
}

extern "C" void kernel_launch(void* const* d_in, const int* in_sizes, int n_in,
                              void* d_out, int out_size, void* d_ws,
                              size_t ws_size, hipStream_t stream) {
  const float* xs     = (const float*)d_in[0];  // [B,128]
  const float* mus    = (const float*)d_in[1];  // [128,1024]
  const float* alphas = (const float*)d_in[2];  // [1024]
  float* out = (float*)d_out;

  _Float16* bt = (_Float16*)d_ws;                       // 256 KB
  float* ea = (float*)((char*)d_ws + (size_t)KPROTO * DDIM * sizeof(_Float16));

  const int B = in_sizes[0] / DDIM;  // 65536
  prep_kernel<<<KPROTO / 256, 256, 0, stream>>>(mus, alphas, bt, ea);
  softmin_wmma<<<B / 128, 256, 0, stream>>>(xs, bt, ea, out);
}
